// MultiHeadDotProductAttention_73839077753280
// MI455X (gfx1250) — matrix operand order repaired and checked by reading, not compiled, not run on hardware
//
#include <hip/hip_runtime.h>
#include <cstdint>

// Problem constants (match the reference).
#define BB 2
#define SS 2048
#define DD 2048
#define HH 16
#define HDD 128
#define MROWS (BB * SS)      // 4096 flattened (b, s) rows

typedef __attribute__((ext_vector_type(16))) __bf16 v16bf;
typedef __attribute__((ext_vector_type(8)))  __bf16 v8bf;
typedef __attribute__((ext_vector_type(8)))  float  v8f;
typedef __attribute__((ext_vector_type(4)))  unsigned int tdm_u32x4;
typedef __attribute__((ext_vector_type(8)))  int          tdm_i32x8;
typedef __attribute__((ext_vector_type(4)))  int          tdm_i32x4;

#if defined(__has_builtin)
#  if __has_builtin(__builtin_amdgcn_tensor_load_to_lds) && __has_builtin(__builtin_amdgcn_s_wait_tensorcnt)
#    define HAVE_TDM 1
#  else
#    define HAVE_TDM 0
#  endif
#else
#  define HAVE_TDM 0
#endif

// Low 32 bits of a flat shared-aperture address == LDS byte offset (ISA 10.2).
__device__ __forceinline__ unsigned lds_offset_of(const void* p) {
  return (unsigned)(uintptr_t)p;
}

__device__ __forceinline__ v16bf cat8(v8bf lo, v8bf hi) {
  return __builtin_shufflevector(lo, hi, 0,1,2,3,4,5,6,7,8,9,10,11,12,13,14,15);
}

__device__ __forceinline__ v8f wmma_bf16(v16bf a, v16bf b, v8f c) {
  // D = A(16x32) * B(32x16) + C, f32 accumulate.
  return __builtin_amdgcn_wmma_f32_16x16x32_bf16(false, a, false, b, (short)0, c,
                                                 false, false);
}

#if HAVE_TDM
// 2D tile load global->LDS via Tensor Data Mover. bf16 elements (data_size=1).
// D# group layouts per CDNA5 ISA.
__device__ __forceinline__ void tdm_load_2d_bf16(unsigned lds_byte,
                                                 const void* gsrc,
                                                 unsigned tile_cols,   // dim0, elems
                                                 unsigned tile_rows,   // dim1, rows
                                                 unsigned rows_avail,  // tensor_dim1
                                                 unsigned row_stride)  // elems
{
  unsigned long long ga = (unsigned long long)(uintptr_t)gsrc;
  tdm_u32x4 g0 = {
      1u,                                             // count=1, user descriptor
      lds_byte,                                       // lds_addr
      (unsigned)(ga & 0xFFFFFFFFull),                 // global_addr lo
      (unsigned)((ga >> 32) & 0x01FFFFFFull) | 0x80000000u };  // addr hi | type=2
  tdm_i32x8 g1 = {
      (int)(1u << 16),                                              // data_size=1 (2B)
      (int)((tile_cols & 0xFFFFu) << 16),                           // tensor_dim0 lo16
      (int)(((tile_cols >> 16) & 0xFFFFu) | ((rows_avail & 0xFFFFu) << 16)),
      (int)(((rows_avail >> 16) & 0xFFFFu) | ((tile_cols & 0xFFFFu) << 16)), // tile_dim0
      (int)(tile_rows & 0xFFFFu),                                   // tile_dim1
      (int)row_stride,                                              // dim0_stride lo32
      0, 0 };
  tdm_i32x4 z4 = {0, 0, 0, 0};
#if defined(__clang_major__) && (__clang_major__ >= 23)
  tdm_i32x8 z8 = {0, 0, 0, 0, 0, 0, 0, 0};
  __builtin_amdgcn_tensor_load_to_lds(g0, g1, z4, z4, z8, 0);
#else
  __builtin_amdgcn_tensor_load_to_lds(g0, g1, z4, z4, 0);
#endif
}
#endif

// ---------------------------------------------------------------------------
// f32 -> bf16 conversion (grid-stride)
// ---------------------------------------------------------------------------
__global__ void cvt_f32_to_bf16_kernel(const float* __restrict__ s,
                                       __bf16* __restrict__ d, long n) {
  long i  = (long)blockIdx.x * blockDim.x + threadIdx.x;
  long st = (long)gridDim.x * blockDim.x;
  for (; i < n; i += st) d[i] = (__bf16)s[i];
}

// ---------------------------------------------------------------------------
// RoPE sin/cos tables, [S][128] f32 (concat-duplicated halves like flaxformer)
// ---------------------------------------------------------------------------
__global__ void rope_tables_kernel(float* __restrict__ sinT, float* __restrict__ cosT) {
  int idx = blockIdx.x * blockDim.x + threadIdx.x;  // SS*64 threads
  if (idx >= SS * 64) return;
  int s = idx >> 6, f = idx & 63;
  float fraction = (2.0f * f) * (1.0f / (float)HDD);
  float inv_ts   = __powf(10000.0f, -fraction);
  float ang      = (float)s * inv_ts;
  float sv = __sinf(ang), cv = __cosf(ang);
  sinT[s * HDD + f] = sv;  sinT[s * HDD + 64 + f] = sv;
  cosT[s * HDD + f] = cv;  cosT[s * HDD + 64 + f] = cv;
}

// ---------------------------------------------------------------------------
// In-place RoPE on [B,S,H,128] bf16 (pair f, f+64), with optional logit scale
// ---------------------------------------------------------------------------
__global__ void rope_apply_kernel(__bf16* __restrict__ X,
                                  const float* __restrict__ sinT,
                                  const float* __restrict__ cosT, float scale) {
  long idx = (long)blockIdx.x * blockDim.x + threadIdx.x;  // BB*SS*HH*64
  if (idx >= (long)BB * SS * HH * 64) return;
  int  f  = (int)(idx & 63);
  long u  = idx >> 6;
  int  hh = (int)(u % HH);
  long bs = u / HH;
  int  s  = (int)(bs % SS);
  size_t base = ((size_t)bs * HH + hh) * HDD;
  float x1 = (float)X[base + f];
  float x2 = (float)X[base + 64 + f];
  float sn = sinT[s * HDD + f], cs = cosT[s * HDD + f];
  X[base + f]      = (__bf16)((x1 * cs - x2 * sn) * scale);
  X[base + 64 + f] = (__bf16)((x2 * cs + x1 * sn) * scale);
}

// ---------------------------------------------------------------------------
// bf16 GEMM: C[M,N] = A[M,K] x B[K,N]; 128x128 tile, 8 waves, WMMA 16x16x32.
// A tile staged via TDM (when available); B tile staged transposed in LDS so
// B-fragments are pairs of contiguous 16B ds loads. Output bf16 (Cb) or f32 (Cf).
// ---------------------------------------------------------------------------
__global__ __launch_bounds__(256) void gemm_bf16(const __bf16* __restrict__ A,
                                                 const __bf16* __restrict__ Bm,
                                                 __bf16* __restrict__ Cb,
                                                 float* __restrict__ Cf,
                                                 int M, int N, int K) {
  __shared__ __bf16 As[2][128 * 32];   // [m][k] row-major
  __shared__ __bf16 Bs[2][128 * 32];   // transposed: [n][k]

  const int t = threadIdx.x;
  const int lane = t & 31, w = t >> 5;
  const int wm = w >> 2, wn = w & 3;          // 2x4 wave grid -> 64x32 per wave
  const int lrow = lane & 15, lhalf = lane >> 4;
  const int m0 = blockIdx.y * 128, n0 = blockIdx.x * 128;
  const int nk = K >> 5;

  const v8f vzero = {0.f, 0.f, 0.f, 0.f, 0.f, 0.f, 0.f, 0.f};
  v8f acc[4][2];
#pragma unroll
  for (int mi = 0; mi < 4; ++mi)
#pragma unroll
    for (int ni = 0; ni < 2; ++ni) acc[mi][ni] = vzero;

  auto loadA = [&](int kt, int buf) {
#if HAVE_TDM
    if (w == 0)
      tdm_load_2d_bf16(lds_offset_of(&As[buf][0]), A + (size_t)m0 * K + kt * 32,
                       32u, 128u, (unsigned)(M - m0), (unsigned)K);
#else
#pragma unroll
    for (int i = 0; i < 2; ++i) {
      int c = t + i * 256;                       // 512 chunks of 8 elems
      int row = c >> 2, seg = c & 3;
      uint4 v = *(const uint4*)(A + (size_t)(m0 + row) * K + kt * 32 + seg * 8);
      *(uint4*)&As[buf][row * 32 + seg * 8] = v;
    }
#endif
  };
  auto loadB = [&](int kt, int buf) {            // store transposed [n][k]
#pragma unroll
    for (int i = 0; i < 2; ++i) {
      int c = t + i * 256;
      int krow = c >> 4, nseg = c & 15;
      uint4 v = *(const uint4*)(Bm + (size_t)(kt * 32 + krow) * N + n0 + nseg * 8);
      const __bf16* pv = (const __bf16*)&v;
#pragma unroll
      for (int j = 0; j < 8; ++j) Bs[buf][(nseg * 8 + j) * 32 + krow] = pv[j];
    }
  };

  loadA(0, 0);
  loadB(0, 0);
#if HAVE_TDM
  if (w == 0) __builtin_amdgcn_s_wait_tensorcnt(0);
#endif
  __syncthreads();

  int buf = 0;
  for (int kt = 0; kt < nk; ++kt) {
    if (kt + 1 < nk) { loadA(kt + 1, buf ^ 1); loadB(kt + 1, buf ^ 1); }

    v16bf afrag[4], bfrag[2];
#pragma unroll
    for (int mi = 0; mi < 4; ++mi) {
      const __bf16* p = &As[buf][(wm * 64 + mi * 16 + lrow) * 32 + lhalf * 8];
      afrag[mi] = cat8(*(const v8bf*)p, *(const v8bf*)(p + 16));
    }
#pragma unroll
    for (int ni = 0; ni < 2; ++ni) {
      const __bf16* p = &Bs[buf][(wn * 32 + ni * 16 + lrow) * 32 + lhalf * 8];
      bfrag[ni] = cat8(*(const v8bf*)p, *(const v8bf*)(p + 16));
    }
#pragma unroll
    for (int mi = 0; mi < 4; ++mi)
#pragma unroll
      for (int ni = 0; ni < 2; ++ni)
        acc[mi][ni] = wmma_bf16(afrag[mi], bfrag[ni], acc[mi][ni]);

#if HAVE_TDM
    if (w == 0 && kt + 1 < nk) __builtin_amdgcn_s_wait_tensorcnt(0);
#endif
    __syncthreads();
    buf ^= 1;
  }

  // C layout: VGPR r holds row r (lanes 0-15) / r+8 (lanes 16-31), col = lane%16
#pragma unroll
  for (int mi = 0; mi < 4; ++mi)
#pragma unroll
    for (int ni = 0; ni < 2; ++ni)
#pragma unroll
      for (int r = 0; r < 8; ++r) {
        size_t row = (size_t)m0 + wm * 64 + mi * 16 + r + lhalf * 8;
        size_t col = (size_t)n0 + wn * 32 + ni * 16 + lrow;
        float v = acc[mi][ni][r];
        if (Cf) Cf[row * N + col] = v;
        else    Cb[row * N + col] = (__bf16)v;
      }
}

// ---------------------------------------------------------------------------
// Flash attention: per (b, h, 128-row q tile). 8 waves x 16-row stripes.
// K tile via TDM; V tile transposed in LDS; online softmax; causal mask only
// on the diagonal k tile. Writes bf16 context [B,S,H,128].
// ---------------------------------------------------------------------------
__global__ __launch_bounds__(256) void flash_attn(const __bf16* __restrict__ Qm,
                                                  const __bf16* __restrict__ Km,
                                                  const __bf16* __restrict__ Vm,
                                                  __bf16* __restrict__ Om) {
  extern __shared__ __bf16 smem[];
  __bf16* Kt = smem;                     // [kpos 128][f 128]
  __bf16* Vt = smem + 128 * 128;         // transposed: [f 128][kpos 128]
  __bf16* Pb = smem + 2 * 128 * 128;     // 8 waves x [row 16][kpos 128]

  const int t = threadIdx.x, lane = t & 31, w = t >> 5;
  const int lrow = lane & 15, lhalf = lane >> 4;
  const int qt = blockIdx.x, h = blockIdx.y, b = blockIdx.z;
  const int q0 = qt * 128;
  const size_t rs = (size_t)HH * HDD;    // 2048 elems between seq rows

  // Q A-fragments for the wave's 16-row stripe (already roped + scaled).
  v16bf qfrag[4];
  {
    const __bf16* qp = Qm + ((size_t)b * SS + q0 + w * 16 + lrow) * rs + (size_t)h * HDD;
#pragma unroll
    for (int kk = 0; kk < 4; ++kk) {
      const __bf16* p = qp + kk * 32 + lhalf * 8;
      qfrag[kk] = cat8(*(const v8bf*)p, *(const v8bf*)(p + 16));
    }
  }

  const v8f vzero = {0.f, 0.f, 0.f, 0.f, 0.f, 0.f, 0.f, 0.f};
  v8f ctx[8];
#pragma unroll
  for (int o = 0; o < 8; ++o) ctx[o] = vzero;
  float mrun[8], lrun[8];
#pragma unroll
  for (int r = 0; r < 8; ++r) { mrun[r] = -1e30f; lrun[r] = 0.f; }

  __bf16* Pw = Pb + w * (16 * 128);      // wave-private P staging

  for (int kt = 0; kt <= qt; ++kt) {
    const int k0 = kt * 128;
    __syncthreads();                     // tiles from previous iter fully consumed

#if HAVE_TDM
    if (w == 0)
      tdm_load_2d_bf16(lds_offset_of(Kt),
                       Km + ((size_t)b * SS + k0) * rs + (size_t)h * HDD,
                       128u, 128u, (unsigned)(SS - k0), (unsigned)rs);
#else
#pragma unroll
    for (int i = 0; i < 8; ++i) {
      int c = t + i * 256;
      int kr = c >> 4, fseg = c & 15;
      uint4 v = *(const uint4*)(Km + ((size_t)b * SS + k0 + kr) * rs +
                                (size_t)h * HDD + fseg * 8);
      *(uint4*)&Kt[kr * 128 + fseg * 8] = v;
    }
#endif
    // V tile, stored transposed [f][kpos] so P*V B-fragments are contiguous.
#pragma unroll
    for (int i = 0; i < 8; ++i) {
      int c = t + i * 256;
      int kr = c >> 4, fseg = c & 15;
      uint4 v = *(const uint4*)(Vm + ((size_t)b * SS + k0 + kr) * rs +
                                (size_t)h * HDD + fseg * 8);
      const __bf16* pv = (const __bf16*)&v;
#pragma unroll
      for (int j = 0; j < 8; ++j) Vt[(fseg * 8 + j) * 128 + kr] = pv[j];
    }
#if HAVE_TDM
    if (w == 0) __builtin_amdgcn_s_wait_tensorcnt(0);
#endif
    __syncthreads();

    // logits S = Q K^T : 16 x 128 per wave (8 n-tiles x 4 k-steps)
    v8f sacc[8];
#pragma unroll
    for (int j = 0; j < 8; ++j) sacc[j] = vzero;
#pragma unroll
    for (int kk = 0; kk < 4; ++kk)
#pragma unroll
      for (int j = 0; j < 8; ++j) {
        const __bf16* p = Kt + (j * 16 + lrow) * 128 + kk * 32 + lhalf * 8;
        sacc[j] = wmma_bf16(qfrag[kk], cat8(*(const v8bf*)p, *(const v8bf*)(p + 16)), sacc[j]);
      }

    // causal mask (only the diagonal k tile can cross the diagonal)
    if (kt == qt) {
#pragma unroll
      for (int j = 0; j < 8; ++j)
#pragma unroll
        for (int r = 0; r < 8; ++r) {
          int kpos = k0 + j * 16 + lrow;
          int qrow = q0 + w * 16 + r + lhalf * 8;
          sacc[j][r] = (kpos > qrow) ? -1e30f : sacc[j][r];
        }
    }

    // online softmax; row stats live per-lane for rows r + 8*(lane/16)
    float mnew[8], alpha[8];
#pragma unroll
    for (int r = 0; r < 8; ++r) {
      float v = -1e30f;
#pragma unroll
      for (int j = 0; j < 8; ++j) v = fmaxf(v, sacc[j][r]);
      v = fmaxf(v, __shfl_xor(v, 1));
      v = fmaxf(v, __shfl_xor(v, 2));
      v = fmaxf(v, __shfl_xor(v, 4));
      v = fmaxf(v, __shfl_xor(v, 8));
      mnew[r]  = fmaxf(mrun[r], v);
      alpha[r] = __expf(mrun[r] - mnew[r]);
      mrun[r]  = mnew[r];
    }
#pragma unroll
    for (int r = 0; r < 8; ++r) {
      float rsum = 0.f;
#pragma unroll
      for (int j = 0; j < 8; ++j) {
        float p = __expf(sacc[j][r] - mnew[r]);
        rsum += p;
        Pw[(r + lhalf * 8) * 128 + j * 16 + lrow] = (__bf16)p;  // C->A reshape via LDS
      }
      rsum += __shfl_xor(rsum, 1);
      rsum += __shfl_xor(rsum, 2);
      rsum += __shfl_xor(rsum, 4);
      rsum += __shfl_xor(rsum, 8);
      lrun[r] = lrun[r] * alpha[r] + rsum;
    }
#pragma unroll
    for (int o = 0; o < 8; ++o)
#pragma unroll
      for (int r = 0; r < 8; ++r) ctx[o][r] *= alpha[r];

    // ctx += P x V (same-wave LDS ops are in-order; Pw is wave-private)
#pragma unroll
    for (int kk = 0; kk < 4; ++kk) {
      const __bf16* pa = Pw + lrow * 128 + kk * 32 + lhalf * 8;
      v16bf pf = cat8(*(const v8bf*)pa, *(const v8bf*)(pa + 16));
#pragma unroll
      for (int o = 0; o < 8; ++o) {
        const __bf16* pb = Vt + (o * 16 + lrow) * 128 + kk * 32 + lhalf * 8;
        ctx[o] = wmma_bf16(pf, cat8(*(const v8bf*)pb, *(const v8bf*)(pb + 16)), ctx[o]);
      }
    }
  }

#pragma unroll
  for (int o = 0; o < 8; ++o)
#pragma unroll
    for (int r = 0; r < 8; ++r) {
      size_t qrow = (size_t)q0 + w * 16 + r + lhalf * 8;
      float v = ctx[o][r] / lrun[r];
      Om[((size_t)b * SS + qrow) * rs + (size_t)h * HDD + o * 16 + lrow] = (__bf16)v;
    }
}

// ---------------------------------------------------------------------------
// Launcher
// ---------------------------------------------------------------------------
extern "C" void kernel_launch(void* const* d_in, const int* in_sizes, int n_in,
                              void* d_out, int out_size, void* d_ws, size_t ws_size,
                              hipStream_t stream) {
  (void)in_sizes; (void)n_in; (void)out_size; (void)ws_size;
  const float* xq  = (const float*)d_in[0];
  const float* xkv = (const float*)d_in[1];
  const float* wq  = (const float*)d_in[2];
  const float* wk  = (const float*)d_in[3];
  const float* wv  = (const float*)d_in[4];
  const float* wo  = (const float*)d_in[5];
  float* out = (float*)d_out;            // d_in[6] = causal mask (handled analytically)

  const long E = (long)BB * SS * DD;     // 8388608
  const long W = (long)DD * HH * HDD;    // 4194304

  char* ws = (char*)d_ws;
  size_t off = 0;
  auto take = [&](size_t bytes) {
    size_t o = off; off += (bytes + 255) & ~(size_t)255; return (void*)(ws + o);
  };
  __bf16* xq_b  = (__bf16*)take((size_t)E * 2);
  __bf16* xkv_b = (__bf16*)take((size_t)E * 2);
  __bf16* wq_b  = (__bf16*)take((size_t)W * 2);
  __bf16* wk_b  = (__bf16*)take((size_t)W * 2);
  __bf16* wv_b  = (__bf16*)take((size_t)W * 2);
  __bf16* wo_b  = (__bf16*)take((size_t)W * 2);
  __bf16* Qb    = (__bf16*)take((size_t)E * 2);
  __bf16* Kb    = (__bf16*)take((size_t)E * 2);
  __bf16* Vb    = (__bf16*)take((size_t)E * 2);
  __bf16* Cb    = (__bf16*)take((size_t)E * 2);
  float*  sinT  = (float*)take((size_t)SS * HDD * 4);
  float*  cosT  = (float*)take((size_t)SS * HDD * 4);

  cvt_f32_to_bf16_kernel<<<2048, 256, 0, stream>>>(xq,  xq_b,  E);
  cvt_f32_to_bf16_kernel<<<2048, 256, 0, stream>>>(xkv, xkv_b, E);
  cvt_f32_to_bf16_kernel<<<1024, 256, 0, stream>>>(wq,  wq_b,  W);
  cvt_f32_to_bf16_kernel<<<1024, 256, 0, stream>>>(wk,  wk_b,  W);
  cvt_f32_to_bf16_kernel<<<1024, 256, 0, stream>>>(wv,  wv_b,  W);
  cvt_f32_to_bf16_kernel<<<1024, 256, 0, stream>>>(wo,  wo_b,  W);
  rope_tables_kernel<<<(SS * 64) / 256, 256, 0, stream>>>(sinT, cosT);

  dim3 gg(DD / 128, MROWS / 128);        // (16, 32) tiles
  gemm_bf16<<<gg, 256, 0, stream>>>(xq_b,  wq_b, Qb, nullptr, MROWS, DD, DD);
  gemm_bf16<<<gg, 256, 0, stream>>>(xkv_b, wk_b, Kb, nullptr, MROWS, DD, DD);
  gemm_bf16<<<gg, 256, 0, stream>>>(xkv_b, wv_b, Vb, nullptr, MROWS, DD, DD);

  const long pairs = (long)BB * SS * HH * 64;
  rope_apply_kernel<<<(unsigned)(pairs / 256), 256, 0, stream>>>(
      Qb, sinT, cosT, 0.088388347648318447f /* 128^-0.5 folded into Q */);
  rope_apply_kernel<<<(unsigned)(pairs / 256), 256, 0, stream>>>(Kb, sinT, cosT, 1.0f);

  dim3 fg(SS / 128, HH, BB);
  flash_attn<<<fg, 256, 3 * 128 * 128 * 2 /* 96 KB dynamic LDS */, stream>>>(Qb, Kb, Vb, Cb);

  gemm_bf16<<<gg, 256, 0, stream>>>(Cb, wo_b, nullptr, out, MROWS, DD, DD);
}
